// _SimpleRoutedExperts_16226386444699
// MI455X (gfx1250) — matrix operand order repaired and checked by reading, not compiled, not run on hardware
//
#include <hip/hip_runtime.h>

// Problem sizes (fixed by the reference)
#define T_TOK   4096
#define D_DIM   1024
#define H_DIM   2048
#define E_EXP   8
#define K_TOP   2
#define NASSIGN (T_TOK * K_TOP)   // 8192 (token, k) assignments

typedef __bf16 bf16_t;
typedef bf16_t v16bf __attribute__((ext_vector_type(16)));
typedef bf16_t v8bf  __attribute__((ext_vector_type(8)));
typedef bf16_t v4bf  __attribute__((ext_vector_type(4)));
typedef float  v8f   __attribute__((ext_vector_type(8)));
typedef int    v4i_t __attribute__((ext_vector_type(4)));

// Workspace layout:
//   ints ws[0..7]  counts | ws[8..15] offsets | ws[16..31] spare
//   ws[32..]       tokArr[NASSIGN], then wtArr[NASSIGN] (float)
//   byte 128K      xbf  [T][D]      bf16   (8 MB)     } only if ws_size
//   + 8 MB         W1bf [E][2H][D]  bf16   (64 MB)    } is large enough
//   + 64 MB        W2bf [E][D][H]   bf16   (32 MB)    }
#define XBF_OFF   (128u * 1024u)
#define W1BF_OFF  (XBF_OFF + (size_t)T_TOK * D_DIM * 2)
#define W2BF_OFF  (W1BF_OFF + (size_t)E_EXP * 2 * H_DIM * D_DIM * 2)
#define WS_NEED   (W2BF_OFF + (size_t)E_EXP * D_DIM * H_DIM * 2)

// ----------------- CDNA5 async global->LDS copy (ASYNCcnt) -----------------
#if defined(__has_builtin)
#if __has_builtin(__builtin_amdgcn_global_load_async_to_lds_b128)
#define HAVE_ASYNC_LDS 1
#endif
#endif
#ifndef HAVE_ASYNC_LDS
#define HAVE_ASYNC_LDS 0
#endif

#if HAVE_ASYNC_LDS
__device__ __forceinline__ void cp16_g2l(const bf16_t* g, bf16_t* l) {
    // 16-byte global -> LDS, no VGPR data path, tracked by ASYNCcnt
    __builtin_amdgcn_global_load_async_to_lds_b128((v4i_t*)g, (v4i_t*)l, 0, 0);
}
__device__ __forceinline__ void async_wait_all() {
#if __has_builtin(__builtin_amdgcn_s_wait_asynccnt)
    __builtin_amdgcn_s_wait_asynccnt(0);
#else
    asm volatile("s_wait_asynccnt 0x0" ::: "memory");
#endif
}
#else
__device__ __forceinline__ void cp16_g2l(const bf16_t* g, bf16_t* l) {
    *(uint4*)l = *(const uint4*)g;           // through-VGPR fallback
}
__device__ __forceinline__ void async_wait_all() {}
#endif

// ---------------------------- small helpers --------------------------------
__device__ __forceinline__ v8bf load8(const float* p) {
    const float4* q = (const float4*)p;
    float4 a = q[0], b = q[1];
    v8bf r;
    r[0] = (bf16_t)a.x; r[1] = (bf16_t)a.y; r[2] = (bf16_t)a.z; r[3] = (bf16_t)a.w;
    r[4] = (bf16_t)b.x; r[5] = (bf16_t)b.y; r[6] = (bf16_t)b.z; r[7] = (bf16_t)b.w;
    return r;
}
__device__ __forceinline__ v16bf cat16(v8bf lo, v8bf hi) {
    return __builtin_shufflevector(lo, hi, 0,1,2,3,4,5,6,7,8,9,10,11,12,13,14,15);
}
// copy 16 contiguous source elements into one LDS fragment row
__device__ __forceinline__ void cp_frag(const bf16_t* g, bf16_t* l) {
    cp16_g2l(g, l); cp16_g2l(g + 8, l + 8);
}
__device__ __forceinline__ void cp_frag(const float* g, bf16_t* l) {
    *(v16bf*)l = cat16(load8(g), load8(g + 8));
    __builtin_prefetch(g + 32, 0, 1);        // fp32 fallback path only
}
// gathered-x staging: two 8-element runs (d, d+16) per fragment
__device__ __forceinline__ void cp_xfrag(const bf16_t* row, bf16_t* l) {
    cp16_g2l(row, l); cp16_g2l(row + 16, l + 8);
}
__device__ __forceinline__ void cp_xfrag(const float* row, bf16_t* l) {
    *(v16bf*)l = cat16(load8(row), load8(row + 16));
}

// ---------------------------- routing --------------------------------------
__global__ void zero_y(float* __restrict__ y) {
    int i = blockIdx.x * blockDim.x + threadIdx.x;
    float4 z = {0.f, 0.f, 0.f, 0.f};
    ((float4*)y)[i] = z;
}

__global__ void init_counts(int* __restrict__ ws) {
    if (threadIdx.x < 32) ws[threadIdx.x] = 0;
}

__global__ void count_assign(const int* __restrict__ idx, int* __restrict__ ws) {
    int i = blockIdx.x * blockDim.x + threadIdx.x;
    if (i < NASSIGN) atomicAdd(&ws[idx[i] & (E_EXP - 1)], 1);
}

__global__ void scan_offsets(int* __restrict__ ws) {
    int run = 0;
    for (int e = 0; e < E_EXP; ++e) { ws[8 + e] = run; run += ws[e]; }
}

// Deterministic stable fill: slot = offs[e] + (#earlier assignments to e).
__global__ void fill_assign(const int* __restrict__ idx, const float* __restrict__ w,
                            int* __restrict__ ws) {
    int i = blockIdx.x * blockDim.x + threadIdx.x;
    if (i >= NASSIGN) return;
    int e = idx[i] & (E_EXP - 1);
    int rank = 0;
    for (int j = 0; j < i; ++j) rank += ((idx[j] & (E_EXP - 1)) == e) ? 1 : 0;
    int pos = ws[8 + e] + rank;
    int*   tokArr = ws + 32;
    float* wtArr  = (float*)(ws + 32 + NASSIGN);
    tokArr[pos] = i / K_TOP;
    wtArr[pos]  = w[i];
}

// ---------------------------- fp32 -> bf16 pre-conversion -------------------
__global__ void cvt_bf16(const float* __restrict__ src, bf16_t* __restrict__ dst) {
    long i = (long)blockIdx.x * blockDim.x + threadIdx.x;   // one float4 each
    float4 v = ((const float4*)src)[i];
    v4bf o;
    o[0] = (bf16_t)v.x; o[1] = (bf16_t)v.y; o[2] = (bf16_t)v.z; o[3] = (bf16_t)v.w;
    ((v4bf*)dst)[i] = o;
}

// ---------------------------------------------------------------------------
// Fused routed GatedMLP tile kernel, templated on the global element type
// (bf16 fast path when workspace holds pre-converted copies, fp32 fallback).
//
// Block = 128 threads (4 wave32). blockIdx.y = expert, blockIdx.x = 16-row
// tile of that expert's gathered segment.
//
// LDS tiles are fragment-linear ([tile][lane][16 bf16]) so each WMMA operand
// is one aligned 32B LDS read. W1/W2 staging is double-buffered and uses the
// CDNA5 async global->LDS path: steady state per K-step is
//   s_wait_asynccnt 0 ; s_barrier ; issue async for next buffer ; ds_load+wmma
// Buffer WAR is safe: operands are dscnt-waited into VGPRs before each wave
// reaches the next barrier, and new async writes are only issued post-barrier.
// ---------------------------------------------------------------------------
template <typename TS>
__launch_bounds__(128)
__global__ void moe_mlp(const TS* __restrict__ xg,
                        const TS* __restrict__ W1,
                        const TS* __restrict__ W2,
                        const int* __restrict__ ws_i,
                        float*     __restrict__ y) {
    const int e    = blockIdx.y;
    const int cnt  = ws_i[e];
    const int off  = ws_i[8 + e];
    const int row0 = blockIdx.x * 16;
    if (row0 >= cnt) return;                 // uniform per block: EXEC stays full
    const int*   tokArr = ws_i + 32;
    const float* wtArr  = (const float*)(ws_i + 32 + NASSIGN);

    // fragment-linear LDS buffers
    __shared__ __align__(32) bf16_t XsF[32 * 32 * 16];     // [d-chunk][lane][16] 32 KB
    __shared__ __align__(32) bf16_t W1F[2][4 * 32 * 16];   // double-buffered, 2x4 KB
    __shared__ __align__(32) bf16_t W2F[2][8 * 32 * 16];   // double-buffered, 2x8 KB
    __shared__ __align__(32) bf16_t AaF[32 * 16];          // activated A2 frag 1 KB
    __shared__ float  UG[16][68];                          // u cols 0..31 | gate 32..63
    __shared__ int    toks[16];
    __shared__ float  wts[16];

    const int tid   = threadIdx.x;
    const int lane  = tid & 31;
    const int wv    = tid >> 5;
    const int ln16  = lane & 15;
    const int lhalf = lane >> 4;

    if (tid < 16) {
        int r = row0 + tid;
        if (r < cnt) { toks[tid] = tokArr[off + r]; wts[tid] = wtArr[off + r]; }
        else         { toks[tid] = -1;              wts[tid] = 0.0f; }
    }
    __syncthreads();

    // Stage gathered x rows once, directly in A-fragment order.
    for (int u = tid; u < 32 * 32; u += 128) {
        int chunk = u >> 5, lp = u & 31;
        int m = lp & 15, half = lp >> 4;
        int t = toks[m]; t = (t < 0) ? 0 : t;    // ragged rows: weight 0 kills them
        cp_frag(xg + (long)t * D_DIM + chunk * 32 + half * 16, XsF + u * 16);
    }

    const TS* W1e = W1 + (long)e * 2 * H_DIM * D_DIM;
    const TS* W2e = W2 + (long)e * D_DIM * H_DIM;

    v8f accOut[16];
#pragma unroll
    for (int a = 0; a < 16; ++a) accOut[a] = (v8f){0,0,0,0,0,0,0,0};

    // thread-constant staging geometry
    const int s_tile = tid >> 5, s_lp = tid & 31;
    const int s_n = s_lp & 15, s_kh = s_lp >> 4;
    const long w1row = (s_tile >= 2 ? H_DIM : 0) + (s_tile & 1) * 16 + s_n;

    async_wait_all();
    __syncthreads();

    const v16bf* XsV = (const v16bf*)XsF;
    const v16bf* AaV = (const v16bf*)AaF;

    for (int hc = 0; hc < H_DIM / 32; ++hc) {
        // ---------------- fc1: [16 x 32] (u|gate) = Xs @ W1e^T ----------------
        // prologue: fill buffer 0 for d0 = 0
        cp_frag(W1e + (w1row + hc * 32) * D_DIM + s_kh * 16, W1F[0] + tid * 16);
        v8f accUG = (v8f){0,0,0,0,0,0,0,0};
        int cur = 0;
        for (int d0 = 0; d0 < D_DIM; d0 += 32, cur ^= 1) {
            async_wait_all();
            __syncthreads();
            if (d0 + 32 < D_DIM)             // issue next tile into idle buffer
                cp_frag(W1e + (w1row + hc * 32) * D_DIM + (d0 + 32) + s_kh * 16,
                        W1F[cur ^ 1] + tid * 16);
            v16bf A = XsV[(d0 >> 5) * 32 + lane];
            v16bf B = ((const v16bf*)W1F[cur])[wv * 32 + lane];
            accUG = __builtin_amdgcn_wmma_f32_16x16x32_bf16(
                false, A, false, B, (short)0, accUG, false, false);
        }

        // prologue for fc2 piece 0 (overlaps UG write + activation)
        {
            long dcol = s_tile * 16 + s_n;   // 2 units per thread
            cp_frag(W2e + dcol * H_DIM + hc * 32 + s_kh * 16, W2F[0] + tid * 16);
            dcol += 64;
            cp_frag(W2e + dcol * H_DIM + hc * 32 + s_kh * 16, W2F[0] + (tid + 128) * 16);
        }

        __syncthreads();                     // UG overwrite protection
#pragma unroll
        for (int r = 0; r < 8; ++r)          // C layout: m = r + 8*lhalf, n = ln16
            UG[r + 8 * lhalf][wv * 16 + ln16] = accUG[r];
        __syncthreads();

        // ---------------- activation: a = u * silu(g), stored in A2-frag order
        for (int i = tid; i < 16 * 32; i += 128) {
            int m = i >> 5, c = i & 31;
            float uu = UG[m][c];
            float g  = UG[m][32 + c];
            float a  = uu * (g / (1.0f + __expf(-g)));
            int half = c >> 4;
            int ii   = c & 15;
            AaF[(m + 16 * half) * 16 + ii] = (bf16_t)a;
        }
        __syncthreads();
        v16bf A2 = AaV[lane];

        // ---------------- fc2: accumulate all 1024 out cols (8 x 128-col pieces)
        int cur2 = 0;
        for (int p = 0; p < 8; ++p, cur2 ^= 1) {
            async_wait_all();
            __syncthreads();
            if (p + 1 < 8) {                 // issue next piece into idle buffer
                long dcol = (p + 1) * 128 + s_tile * 16 + s_n;
                cp_frag(W2e + dcol * H_DIM + hc * 32 + s_kh * 16,
                        W2F[cur2 ^ 1] + tid * 16);
                dcol += 64;
                cp_frag(W2e + dcol * H_DIM + hc * 32 + s_kh * 16,
                        W2F[cur2 ^ 1] + (tid + 128) * 16);
            }
#pragma unroll
            for (int q = 0; q < 2; ++q) {    // wave wv owns tiles wv, wv+4
                v16bf B2 = ((const v16bf*)W2F[cur2])[(wv + 4 * q) * 32 + lane];
                accOut[p * 2 + q] = __builtin_amdgcn_wmma_f32_16x16x32_bf16(
                    false, A2, false, B2, (short)0, accOut[p * 2 + q], false, false);
            }
        }
    }

    // ---------------- weighted scatter-add into y ----------------
#pragma unroll
    for (int a = 0; a < 16; ++a) {
        int p = a >> 1, q = a & 1;
        int col = p * 128 + (wv + 4 * q) * 16 + ln16;
#pragma unroll
        for (int r = 0; r < 8; ++r) {
            int m = r + 8 * lhalf;
            int t = toks[m];
            if (t >= 0) {
                __hip_atomic_fetch_add(&y[(long)t * D_DIM + col],
                                       accOut[a][r] * wts[m],
                                       __ATOMIC_RELAXED, __HIP_MEMORY_SCOPE_AGENT);
            }
        }
    }
}

extern "C" void kernel_launch(void* const* d_in, const int* in_sizes, int n_in,
                              void* d_out, int out_size, void* d_ws, size_t ws_size,
                              hipStream_t stream) {
    (void)in_sizes; (void)n_in; (void)out_size;
    const float* x   = (const float*)d_in[0];
    const float* w   = (const float*)d_in[1];
    const int*   idx = (const int*)  d_in[2];
    const float* W1  = (const float*)d_in[3];
    const float* W2  = (const float*)d_in[4];
    float* y = (float*)d_out;
    int*   ws = (int*)d_ws;

    zero_y      <<<(T_TOK * D_DIM / 4) / 256, 256, 0, stream>>>(y);
    init_counts <<<1, 32, 0, stream>>>(ws);
    count_assign<<<(NASSIGN + 255) / 256, 256, 0, stream>>>(idx, ws);
    scan_offsets<<<1, 1, 0, stream>>>(ws);
    fill_assign <<<(NASSIGN + 255) / 256, 256, 0, stream>>>(idx, w, ws);

    if (ws_size >= WS_NEED) {
        // One-time fp32->bf16 conversion: halves weight traffic, strips all
        // converts out of the GEMM hot loops, and enables raw async LDS copies.
        bf16_t* xbf  = (bf16_t*)((char*)d_ws + XBF_OFF);
        bf16_t* w1bf = (bf16_t*)((char*)d_ws + W1BF_OFF);
        bf16_t* w2bf = (bf16_t*)((char*)d_ws + W2BF_OFF);
        cvt_bf16<<<(T_TOK * (D_DIM / 4)) / 256,             256, 0, stream>>>(x,  xbf);
        cvt_bf16<<<(E_EXP * 2 * H_DIM * (D_DIM / 4)) / 256, 256, 0, stream>>>(W1, w1bf);
        cvt_bf16<<<(E_EXP * D_DIM * (H_DIM / 4)) / 256,     256, 0, stream>>>(W2, w2bf);
        moe_mlp<bf16_t><<<dim3(512, E_EXP), 128, 0, stream>>>(xbf, w1bf, w2bf, ws, y);
    } else {
        moe_mlp<float> <<<dim3(512, E_EXP), 128, 0, stream>>>(x, W1, W2, ws, y);
    }
}
